// MultiHeadAttention_16561393893759
// MI455X (gfx1250) — matrix operand order repaired and checked by reading, not compiled, not run on hardware
//
#include <hip/hip_runtime.h>
#include <hip/hip_bf16.h>
#include <hip/hip_fp16.h>

typedef __attribute__((ext_vector_type(16))) _Float16 v16h;
typedef __attribute__((ext_vector_type(8)))  _Float16 v8h;
typedef __attribute__((ext_vector_type(8)))  float    v8f;
typedef __attribute__((ext_vector_type(4)))  unsigned v4u;
typedef __attribute__((ext_vector_type(8)))  unsigned v8u;

#define NEMBD 384
#define NHEAD 6
#define HS    64
#define TSEQ  128
#define NB    512

// ---------------------------------------------------------------------------
// WMMA helpers (CDNA5 v_wmma_f32_16x16x32_f16, wave32)
// A operand (16x32 f16): lane&15 = row M, group g=lane>>4:
//   elems 0..7  = K[g*8 .. g*8+7], elems 8..15 = K[g*8+16 .. g*8+23]
// B operand (32x16 f16): lane&15 = col N, group g:
//   elems 0..7 = K[g*8 .. g*8+7], elems 8..15 = K[g*8+16 .. g*8+23]  (B fed from a [n][k] transposed view)
// C/D (16x16 f32): lane&15 = N, VGPR i -> M = (lane>>4)*8 + i
// ---------------------------------------------------------------------------
static __device__ __forceinline__ v16h cat16(v8h lo, v8h hi) {
  v16h r;
#pragma unroll
  for (int i = 0; i < 8; ++i) { r[i] = lo[i]; r[i + 8] = hi[i]; }
  return r;
}

static __device__ __forceinline__ v16h load_a16(const _Float16* tile, int ld, int lane) {
  const int r = lane & 15, g = lane >> 4;
  const _Float16* p = tile + r * ld + g * 8;
  return cat16(*(const v8h*)p, *(const v8h*)(p + 16));
}

static __device__ __forceinline__ v16h load_bT(const _Float16* tileT, int ld, int lane) {
  const int n = lane & 15, g = lane >> 4;
  const _Float16* p = tileT + n * ld + g * 8;
  return cat16(*(const v8h*)p, *(const v8h*)(p + 16));
}

static __device__ __forceinline__ v8f wmma16(v16h a, v16h b, v8f c) {
  return __builtin_amdgcn_wmma_f32_16x16x32_f16(false, a, false, b, (short)0, c,
                                                false, false);
}

static __device__ __forceinline__ v8f vzero8() {
  return (v8f){0.f, 0.f, 0.f, 0.f, 0.f, 0.f, 0.f, 0.f};
}

// Tensor Data Mover: issue a 1-D tile copy global->LDS described by D# groups.
static __device__ __forceinline__ void tdm_load_issue(v4u g0, v8u g1) {
  asm volatile("tensor_load_to_lds %0, %1" ::"s"(g0), "s"(g1) : "memory");
}

// Async per-lane 16B copy global->LDS (ASYNCcnt tracked).
static __device__ __forceinline__ void async_copy_b128(unsigned lds_off,
                                                       const void* gaddr) {
  asm volatile("global_load_async_to_lds_b128 %0, %1, off" ::"v"(lds_off),
               "v"(gaddr)
               : "memory");
}
static __device__ __forceinline__ void wait_asynccnt0() {
  asm volatile("s_wait_asynccnt 0x0" ::: "memory");
}

// ---------------------------------------------------------------------------
// Kernel 0a: Wq/Wk/Wv f32 [H][C][D] -> f16 transposed [H][3][D][C]
// ---------------------------------------------------------------------------
__global__ __launch_bounds__(256) void prep_qkv_kernel(
    const float* __restrict__ Wq, const float* __restrict__ Wk,
    const float* __restrict__ Wv, _Float16* __restrict__ wt) {
  int tid = blockIdx.x * 256 + threadIdx.x;  // H*C*D = 147456
  if (tid >= NHEAD * NEMBD * HS) return;
  int h = tid / (NEMBD * HS);
  int r = tid % (NEMBD * HS);
  int c = r / HS;
  int d = r % HS;
  size_t dst = (((size_t)h * 3 + 0) * HS + d) * NEMBD + c;
  wt[dst]                          = (_Float16)Wq[tid];
  wt[dst + (size_t)HS * NEMBD]     = (_Float16)Wk[tid];
  wt[dst + (size_t)2 * HS * NEMBD] = (_Float16)Wv[tid];
}

// Kernel 0b: Wo f32 [N][C] -> f16 (row n already K-contiguous for out@Wo.T)
__global__ __launch_bounds__(256) void prep_wo_kernel(
    const float* __restrict__ Wo, _Float16* __restrict__ wo16) {
  int tid = blockIdx.x * 256 + threadIdx.x;
  if (tid >= NEMBD * NEMBD) return;
  wo16[tid] = (_Float16)Wo[tid];
}

// ---------------------------------------------------------------------------
// Kernel 1: LayerNorm, one wave32 per row of 384, output f16 h
// ---------------------------------------------------------------------------
__global__ __launch_bounds__(256) void ln_kernel(
    const float* __restrict__ x, const float* __restrict__ gamma,
    const float* __restrict__ beta, _Float16* __restrict__ h) {
  const int row  = blockIdx.x * 8 + (threadIdx.x >> 5);
  const int lane = threadIdx.x & 31;
  const float* xr = x + (size_t)row * NEMBD;
  float v[12];
  float s = 0.f;
#pragma unroll
  for (int i = 0; i < 12; ++i) { v[i] = xr[lane + i * 32]; s += v[i]; }
#pragma unroll
  for (int d = 1; d < 32; d <<= 1) s += __shfl_xor(s, d, 32);
  const float mu = s * (1.0f / NEMBD);
  float ss = 0.f;
#pragma unroll
  for (int i = 0; i < 12; ++i) { float dl = v[i] - mu; ss += dl * dl; }
#pragma unroll
  for (int d = 1; d < 32; d <<= 1) ss += __shfl_xor(ss, d, 32);
  const float rstd = rsqrtf(ss * (1.0f / NEMBD) + 1e-5f);
  _Float16* hr = h + (size_t)row * NEMBD;
#pragma unroll
  for (int i = 0; i < 12; ++i) {
    int c = lane + i * 32;
    hr[c] = (_Float16)((v[i] - mu) * rstd * gamma[c] + beta[c]);
  }
}

// ---------------------------------------------------------------------------
// Kernel 2: fused per-(batch,head) attention, 8 waves, weights TDM'd to LDS.
// ---------------------------------------------------------------------------
__global__ __launch_bounds__(256) void attn_kernel(
    const _Float16* __restrict__ h,     // [B][T][384]
    const _Float16* __restrict__ wqkv,  // [H][3][64][384] (transposed)
    _Float16* __restrict__ out) {       // [B][T][H*64]
  const int b    = blockIdx.x / NHEAD;
  const int head = blockIdx.x % NHEAD;
  const int lane = threadIdx.x & 31;
  const int wave = threadIdx.x >> 5;  // 0..7 == M tile

  __shared__ __align__(16) _Float16 smemW[3 * HS * NEMBD];  // 144KB weights
  __shared__ __align__(16) _Float16 Qlds[TSEQ][HS];         // 16KB [t][d]
  __shared__ __align__(16) _Float16 Klds[TSEQ][HS];         // 16KB [s][d]
  __shared__ __align__(16) _Float16 Vt[HS][TSEQ];           // 16KB [d][s]
  // P (32KB) aliases the weight region once weights are consumed (barrier-
  // separated), keeping total LDS at 192KB.
  _Float16(*Plds)[TSEQ] = (_Float16(*)[TSEQ])smemW;

  const _Float16* hB = h + (size_t)b * TSEQ * NEMBD;
  const _Float16* wH = wqkv + (size_t)head * 3 * HS * NEMBD;

  const int nloc  = lane & 15;
  const int mbase = (lane >> 4) << 3;  // 0 or 8

  // ---- TDM: each wave DMA's its 1/8 slice of the 3x64x384 f16 weights -----
  {
    const unsigned SLICE_H = 3 * HS * NEMBD / 8;  // halves per wave = 9216
    const unsigned ELEM8   = SLICE_H * 2 / 8;     // 8B elems = 2304 (fits u16)
    unsigned ldsoff = ((unsigned)(uintptr_t)&smemW[0]) + wave * SLICE_H * 2;
    unsigned long long ga =
        (unsigned long long)(uintptr_t)wH + (unsigned long long)wave * SLICE_H * 2;
    v4u g0;
    g0[0] = 1u;  // count=1, user mode
    g0[1] = ldsoff;
    g0[2] = (unsigned)ga;
    g0[3] = (unsigned)((ga >> 32) & 0x1FFFFFFu) | (2u << 30);  // type=2
    v8u g1;
    g1[0] = 3u << 16;                   // data_size = 8B
    g1[1] = (ELEM8 & 0xFFFFu) << 16;    // tensor_dim0 lo16
    g1[2] = (ELEM8 >> 16) | (1u << 16); // tensor_dim0 hi16 | tensor_dim1=1
    g1[3] = (ELEM8 & 0xFFFFu) << 16;    // tile_dim0
    g1[4] = 0u;                         // tile_dim1/2 unused
    g1[5] = ELEM8;                      // tensor_dim0_stride lo32
    g1[6] = 0u;
    g1[7] = 0u;
    tdm_load_issue(g0, g1);
  }

  // ---- Preload this wave's A rows (overlaps the TDM) ----------------------
  v16h aReg[12];
#pragma unroll
  for (int ks = 0; ks < 12; ++ks)
    aReg[ks] = load_a16(hB + (size_t)wave * 16 * NEMBD + ks * 32, NEMBD, lane);

  __builtin_amdgcn_s_wait_tensorcnt(0);
  __syncthreads();

  // ---- Phase 1: QKV projections (A in regs, B from LDS) -------------------
#pragma unroll 1
  for (int nt = 0; nt < 4; ++nt) {
    v8f aq = vzero8(), ak = vzero8(), av = vzero8();
#pragma unroll
    for (int ks = 0; ks < 12; ++ks) {
      v16h bq = load_bT(&smemW[(size_t)(0 * HS + nt * 16) * NEMBD + ks * 32],
                        NEMBD, lane);
      aq = wmma16(aReg[ks], bq, aq);
      v16h bk = load_bT(&smemW[(size_t)(1 * HS + nt * 16) * NEMBD + ks * 32],
                        NEMBD, lane);
      ak = wmma16(aReg[ks], bk, ak);
      v16h bv = load_bT(&smemW[(size_t)(2 * HS + nt * 16) * NEMBD + ks * 32],
                        NEMBD, lane);
      av = wmma16(aReg[ks], bv, av);
    }
    const int d = nt * 16 + nloc;
#pragma unroll
    for (int i = 0; i < 8; ++i) {
      const int m = wave * 16 + mbase + i;
      Qlds[m][d] = (_Float16)aq[i];
      Klds[m][d] = (_Float16)ak[i];
    }
    v8h pk;
#pragma unroll
    for (int i = 0; i < 8; ++i) pk[i] = (_Float16)av[i];
    *(v8h*)&Vt[d][wave * 16 + mbase] = pk;  // contiguous 16B store
  }
  __syncthreads();

  // ---- Phase 2: scores + causal softmax (fully in registers) --------------
  const float scale = 0.125f;  // 1/sqrt(64)
  // Q A-operands are j-invariant: load once.
  v16h qa[2];
#pragma unroll
  for (int ks = 0; ks < 2; ++ks)
    qa[ks] = load_a16(&Qlds[wave * 16][ks * 32], HS, lane);

  v8f S[8];
#pragma unroll
  for (int j = 0; j < 8; ++j) {  // N tiles over s
    v8f acc = vzero8();
#pragma unroll
    for (int ks = 0; ks < 2; ++ks) {
      v16h bb = load_bT(&Klds[j * 16][ks * 32], HS, lane);
      acc = wmma16(qa[ks], bb, acc);
    }
    const int s = j * 16 + nloc;
#pragma unroll
    for (int i = 0; i < 8; ++i) {
      const int t = wave * 16 + mbase + i;
      S[j][i] = (s <= t) ? acc[i] * scale : -__builtin_inff();
    }
  }
  // Row m lives in one 16-lane group -> xor-reduce 1,2,4,8.
  v8f mx = S[0];
#pragma unroll
  for (int j = 1; j < 8; ++j)
#pragma unroll
    for (int i = 0; i < 8; ++i) mx[i] = fmaxf(mx[i], S[j][i]);
#pragma unroll
  for (int d = 1; d < 16; d <<= 1)
#pragma unroll
    for (int i = 0; i < 8; ++i) mx[i] = fmaxf(mx[i], __shfl_xor(mx[i], d, 32));
  v8f sum = vzero8();
#pragma unroll
  for (int j = 0; j < 8; ++j)
#pragma unroll
    for (int i = 0; i < 8; ++i) {
      float e = __expf(S[j][i] - mx[i]);
      S[j][i] = e;
      sum[i] += e;
    }
#pragma unroll
  for (int d = 1; d < 16; d <<= 1)
#pragma unroll
    for (int i = 0; i < 8; ++i) sum[i] += __shfl_xor(sum[i], d, 32);
  v8f rinv;
#pragma unroll
  for (int i = 0; i < 8; ++i) rinv[i] = 1.0f / sum[i];
#pragma unroll
  for (int j = 0; j < 8; ++j) {
    const int s = j * 16 + nloc;
#pragma unroll
    for (int i = 0; i < 8; ++i) {
      const int t = wave * 16 + mbase + i;
      Plds[t][s] = (_Float16)(S[j][i] * rinv[i]);
    }
  }
  __syncthreads();

  // ---- Phase 3: O = P x V --------------------------------------------------
  v8f accO[4];
#pragma unroll
  for (int nt = 0; nt < 4; ++nt) accO[nt] = vzero8();
#pragma unroll
  for (int ks = 0; ks < TSEQ / 32; ++ks) {
    v16h a = load_a16(&Plds[wave * 16][ks * 32], TSEQ, lane);
#pragma unroll
    for (int nt = 0; nt < 4; ++nt) {
      v16h bb = load_bT(&Vt[nt * 16][ks * 32], TSEQ, lane);
      accO[nt] = wmma16(a, bb, accO[nt]);
    }
  }
  _Float16* ob = out + (size_t)b * TSEQ * NEMBD + head * HS;
#pragma unroll
  for (int nt = 0; nt < 4; ++nt) {
    const int d = nt * 16 + nloc;
#pragma unroll
    for (int i = 0; i < 8; ++i) {
      const int t = wave * 16 + mbase + i;
      ob[(size_t)t * NEMBD + d] = (_Float16)accO[nt][i];
    }
  }
}

// ---------------------------------------------------------------------------
// Kernel 3: out = concat[B*T,384] x Wo^T + bo (f32 out); Wo async-staged LDS.
// ---------------------------------------------------------------------------
__global__ __launch_bounds__(256) void proj_kernel(
    const _Float16* __restrict__ att,  // [B*T][384]
    const _Float16* __restrict__ wo,   // [384][384] row n, c contiguous
    const float* __restrict__ bo, float* __restrict__ out) {
  const int b     = blockIdx.x;
  const int lane  = threadIdx.x & 31;
  const int wave  = threadIdx.x >> 5;
  const int nloc  = lane & 15;
  const int mbase = (lane >> 4) << 3;

  __shared__ __align__(16) _Float16 WoLds[NEMBD * NEMBD];  // 288KB

  // Async-stage Wo into LDS (72 x 16B per thread), overlapped with A preload.
  {
    const unsigned base = (unsigned)(uintptr_t)&WoLds[0];
#pragma unroll 1
    for (int it = 0; it < 72; ++it) {
      const int idx = (it * 256 + (int)threadIdx.x) * 8;  // halves
      async_copy_b128(base + (unsigned)idx * 2, wo + idx);
    }
  }

  const _Float16* aB =
      att + (size_t)b * TSEQ * NEMBD + (size_t)wave * 16 * NEMBD;
  float* ob = out + (size_t)b * TSEQ * NEMBD;

  v16h aReg[12];
#pragma unroll
  for (int ks = 0; ks < 12; ++ks) aReg[ks] = load_a16(aB + ks * 32, NEMBD, lane);

  wait_asynccnt0();
  __syncthreads();

#pragma unroll 1
  for (int nt = 0; nt < 24; ++nt) {
    v8f acc = vzero8();
#pragma unroll
    for (int ks = 0; ks < 12; ++ks) {
      v16h bb =
          load_bT(&WoLds[(size_t)(nt * 16) * NEMBD + ks * 32], NEMBD, lane);
      acc = wmma16(aReg[ks], bb, acc);
    }
    const int n = nt * 16 + nloc;
    const float bias = bo[n];
#pragma unroll
    for (int i = 0; i < 8; ++i) {
      const int t = wave * 16 + mbase + i;
      ob[(size_t)t * NEMBD + n] = acc[i] + bias;
    }
  }
}

// ---------------------------------------------------------------------------
extern "C" void kernel_launch(void* const* d_in, const int* in_sizes, int n_in,
                              void* d_out, int out_size, void* d_ws,
                              size_t ws_size, hipStream_t stream) {
  const float* x     = (const float*)d_in[0];
  const float* Wq    = (const float*)d_in[1];
  const float* Wk    = (const float*)d_in[2];
  const float* Wv    = (const float*)d_in[3];
  const float* Wo    = (const float*)d_in[4];
  const float* bo    = (const float*)d_in[5];
  const float* gamma = (const float*)d_in[6];
  const float* beta  = (const float*)d_in[7];
  float* out = (float*)d_out;

  _Float16* ws_h    = (_Float16*)d_ws;                           // 50.3 MB
  _Float16* ws_attn = ws_h + (size_t)NB * TSEQ * NEMBD;          // 50.3 MB
  _Float16* ws_wqkv = ws_attn + (size_t)NB * TSEQ * NEMBD;       // 0.9 MB
  _Float16* ws_wo   = ws_wqkv + (size_t)NHEAD * 3 * HS * NEMBD;  // 0.3 MB

  prep_qkv_kernel<<<(NHEAD * NEMBD * HS + 255) / 256, 256, 0, stream>>>(
      Wq, Wk, Wv, ws_wqkv);
  prep_wo_kernel<<<(NEMBD * NEMBD + 255) / 256, 256, 0, stream>>>(Wo, ws_wo);
  ln_kernel<<<NB * TSEQ / 8, 256, 0, stream>>>(x, gamma, beta, ws_h);
  attn_kernel<<<NB * NHEAD, 256, 0, stream>>>(ws_h, ws_wqkv, ws_attn);
  proj_kernel<<<NB, 256, 0, stream>>>(ws_attn, ws_wo, bo, out);
}
